// GNNClassifier_70153995813360
// MI455X (gfx1250) — matrix operand order repaired and checked by reading, not compiled, not run on hardware
//
#include <hip/hip_runtime.h>
#include <hip/hip_bf16.h>
#include <stdint.h>

// Problem dims (compile-time constants matching the reference)
#define GB      64
#define NNODE   512
#define NEDGE   4096
#define EMBED   512
#define HIDDEN  1024
#define NCLS    16
#define MROWS   (GB * NNODE)          // 32768
#define KTOT    (EMBED + HIDDEN)      // 1536

typedef __attribute__((ext_vector_type(16))) __bf16 v16bf;
typedef __attribute__((ext_vector_type(8)))  float  v8f;

// f32 -> bf16 round-to-nearest-even (16-bit pattern in low bits)
__device__ __forceinline__ unsigned f2bf1(float f) {
    unsigned u = __float_as_uint(f);
    return (u + 0x7FFFu + ((u >> 16) & 1u)) >> 16;
}
__device__ __forceinline__ unsigned f2bf2(float lo, float hi) {
    return f2bf1(lo) | (f2bf1(hi) << 16);
}

union FragU { uint4 q[2]; v16bf v; };   // 32 bytes = 8 VGPRs

#define LDSTRIDE 40   // 32 bf16 payload + 8 pad (80B rows; 16B aligned)

// ---- CDNA5 async-to-LDS copy (ASYNCcnt) -------------------------------
__device__ __forceinline__ void async_copy_b128(const void* gaddr, unsigned ldsaddr) {
    asm volatile("global_load_async_to_lds_b128 %0, %1, off"
                 :: "v"(ldsaddr), "v"(gaddr) : "memory");
}
__device__ __forceinline__ void wait_async0() {
    asm volatile("s_wait_asynccnt 0" ::: "memory");
}
// ---- CDNA5 packed bf16 float atomic -----------------------------------
__device__ __forceinline__ void atomic_pk_add_bf16(void* addr, unsigned data) {
    asm volatile("global_atomic_pk_add_bf16 %0, %1, off"
                 :: "v"(addr), "v"(data) : "memory");
}

// ---------------------------------------------------------------- embedding (gather -> bf16)
__global__ __launch_bounds__(128) void k_embed(const int* __restrict__ tokens,
                                               const float* __restrict__ emb,
                                               unsigned short* __restrict__ hbf) {
    int row = blockIdx.x;                       // node row 0..32767
    int tok = tokens[row];
    const float4* src = (const float4*)(emb + (size_t)tok * EMBED);
    uint2*        dst = (uint2*)(hbf + (size_t)row * EMBED);
    float4 v = src[threadIdx.x];                // 128 thr * 4 = 512 ch
    dst[threadIdx.x] = make_uint2(f2bf2(v.x, v.y), f2bf2(v.z, v.w));
}

// ---------------------------------------------------------------- f32 -> bf16 weight convert
__global__ __launch_bounds__(256) void k_cvt_bf16(const float* __restrict__ src,
                                                  unsigned short* __restrict__ dst,
                                                  int n4) {
    int i = blockIdx.x * 256 + threadIdx.x;
    if (i < n4) {
        float4 v = ((const float4*)src)[i];
        ((uint2*)dst)[i] = make_uint2(f2bf2(v.x, v.y), f2bf2(v.z, v.w));
    }
}

// ---------------------------------------------------------------- GEMM1: hm(f32) = h * Wmsg^T
__global__ __launch_bounds__(256) void k_gemm1(const unsigned short* __restrict__ hbf,
                                               const unsigned short* __restrict__ wbf,
                                               float* __restrict__ hm) {
    __shared__ __align__(16) unsigned short smA[128 * LDSTRIDE];
    __shared__ __align__(16) unsigned short smB[128 * LDSTRIDE];

    const int tid   = threadIdx.x;
    const int lane  = tid & 31;
    const int wave  = tid >> 5;
    const int waveM = wave & 3;     // 4 M groups of 32 rows
    const int waveN = wave >> 2;    // 2 N groups of 64 cols
    const int half  = lane >> 4;
    const int l16   = lane & 15;
    const int mBase = blockIdx.y * 128;
    const int nBase = blockIdx.x * 128;

    const unsigned ldsA0 = (unsigned)(size_t)&smA[0];
    const unsigned ldsB0 = (unsigned)(size_t)&smB[0];

    v8f acc[2][4];
    #pragma unroll
    for (int i = 0; i < 2; ++i)
        #pragma unroll
        for (int j = 0; j < 4; ++j)
            acc[i][j] = (v8f){0.f,0.f,0.f,0.f,0.f,0.f,0.f,0.f};

    for (int k0 = 0; k0 < EMBED; k0 += 32) {
        // stage 128x32 bf16 A and B tiles via async-to-LDS (no VGPR bounce)
        #pragma unroll
        for (int c = 0; c < 2; ++c) {
            int chunk = tid + c * 256;          // 512 chunks of 8 bf16 = 16B
            int row   = chunk >> 2;
            int cseg  = chunk & 3;
            unsigned loff = (unsigned)(row * LDSTRIDE + cseg * 8) * 2u;
            async_copy_b128(hbf + (size_t)(mBase + row) * EMBED + k0 + cseg * 8, ldsA0 + loff);
            async_copy_b128(wbf + (size_t)(nBase + row) * EMBED + k0 + cseg * 8, ldsB0 + loff);
        }
        wait_async0();
        __syncthreads();

        FragU fa[2], fb[4];
        #pragma unroll
        for (int i = 0; i < 2; ++i) {
            int mrow = waveM * 32 + i * 16 + l16;
            fa[i].q[0] = *(const uint4*)&smA[mrow * LDSTRIDE + 8 * half];       // K = 8h..8h+7
            fa[i].q[1] = *(const uint4*)&smA[mrow * LDSTRIDE + 16 + 8 * half];  // K = 16+8h..
        }
        #pragma unroll
        for (int j = 0; j < 4; ++j) {
            int nrow = waveN * 64 + j * 16 + l16;
            fb[j].q[0] = *(const uint4*)&smB[nrow * LDSTRIDE + 8 * half];       // K = 8h..8h+7
            fb[j].q[1] = *(const uint4*)&smB[nrow * LDSTRIDE + 16 + 8 * half];
        }
        #pragma unroll
        for (int i = 0; i < 2; ++i)
            #pragma unroll
            for (int j = 0; j < 4; ++j)
                acc[i][j] = __builtin_amdgcn_wmma_f32_16x16x32_bf16(
                    false, fa[i].v, false, fb[j].v, (short)0, acc[i][j], false, false);
        __syncthreads();
    }

    // store f32 messages
    #pragma unroll
    for (int i = 0; i < 2; ++i) {
        int row0 = mBase + waveM * 32 + i * 16 + 8 * half;
        #pragma unroll
        for (int j = 0; j < 4; ++j) {
            int col = nBase + waveN * 64 + j * 16 + l16;
            #pragma unroll
            for (int v = 0; v < 8; ++v)
                hm[(size_t)(row0 + v) * HIDDEN + col] = acc[i][j][v];
        }
    }
}

// ---------------------------------------------------------------- edge scatter-add (packed bf16 atomics)
__global__ __launch_bounds__(256) void k_scatter(const float* __restrict__ hm,
                                                 const int* __restrict__ edges,
                                                 unsigned short* __restrict__ agg) {
    int eid = blockIdx.x;                 // 0 .. GB*NEDGE-1
    int b = eid / NEDGE;
    int e = eid - b * NEDGE;
    int src = edges[(size_t)b * 2 * NEDGE + e];
    int dst = edges[(size_t)b * 2 * NEDGE + NEDGE + e];
    const float4* srow = (const float4*)(hm + ((size_t)b * NNODE + src) * HIDDEN);
    unsigned short* drow = agg + ((size_t)b * NNODE + dst) * HIDDEN;
    float4 v = srow[threadIdx.x];         // 256 thr * 4 = 1024 ch
    unsigned short* d = drow + threadIdx.x * 4;
    atomic_pk_add_bf16(d,     f2bf2(v.x, v.y));
    atomic_pk_add_bf16(d + 2, f2bf2(v.z, v.w));
}

// ---------------------------------------------------------------- GEMM2: relu([h,agg]*Wupd^T + b) -> pooled sums
__global__ __launch_bounds__(256) void k_gemm2(const unsigned short* __restrict__ hbf,
                                               const unsigned short* __restrict__ agg,
                                               const unsigned short* __restrict__ wbf,
                                               const float* __restrict__ bupd,
                                               float* __restrict__ pooled) {
    __shared__ __align__(16) unsigned short smA[128 * LDSTRIDE];
    __shared__ __align__(16) unsigned short smB[128 * LDSTRIDE];

    const int tid   = threadIdx.x;
    const int lane  = tid & 31;
    const int wave  = tid >> 5;
    const int waveM = wave & 3;
    const int waveN = wave >> 2;
    const int half  = lane >> 4;
    const int l16   = lane & 15;
    const int mBase = blockIdx.y * 128;
    const int nBase = blockIdx.x * 128;

    const unsigned ldsA0 = (unsigned)(size_t)&smA[0];
    const unsigned ldsB0 = (unsigned)(size_t)&smB[0];

    v8f acc[2][4];
    #pragma unroll
    for (int i = 0; i < 2; ++i)
        #pragma unroll
        for (int j = 0; j < 4; ++j)
            acc[i][j] = (v8f){0.f,0.f,0.f,0.f,0.f,0.f,0.f,0.f};

    for (int k0 = 0; k0 < KTOT; k0 += 32) {
        #pragma unroll
        for (int c = 0; c < 2; ++c) {
            int chunk = tid + c * 256;
            int row   = chunk >> 2;
            int cseg  = chunk & 3;
            int kk    = k0 + cseg * 8;
            unsigned loff = (unsigned)(row * LDSTRIDE + cseg * 8) * 2u;
            const unsigned short* pa;
            if (kk < EMBED)
                pa = hbf + (size_t)(mBase + row) * EMBED + kk;
            else
                pa = agg + (size_t)(mBase + row) * HIDDEN + (kk - EMBED);
            async_copy_b128(pa, ldsA0 + loff);
            async_copy_b128(wbf + (size_t)(nBase + row) * KTOT + kk, ldsB0 + loff);
        }
        wait_async0();
        __syncthreads();

        FragU fa[2], fb[4];
        #pragma unroll
        for (int i = 0; i < 2; ++i) {
            int mrow = waveM * 32 + i * 16 + l16;
            fa[i].q[0] = *(const uint4*)&smA[mrow * LDSTRIDE + 8 * half];
            fa[i].q[1] = *(const uint4*)&smA[mrow * LDSTRIDE + 16 + 8 * half];
        }
        #pragma unroll
        for (int j = 0; j < 4; ++j) {
            int nrow = waveN * 64 + j * 16 + l16;
            fb[j].q[0] = *(const uint4*)&smB[nrow * LDSTRIDE + 8 * half];
            fb[j].q[1] = *(const uint4*)&smB[nrow * LDSTRIDE + 16 + 8 * half];
        }
        #pragma unroll
        for (int i = 0; i < 2; ++i)
            #pragma unroll
            for (int j = 0; j < 4; ++j)
                acc[i][j] = __builtin_amdgcn_wmma_f32_16x16x32_bf16(
                    false, fa[i].v, false, fb[j].v, (short)0, acc[i][j], false, false);
        __syncthreads();
    }

    // epilogue: bias + relu + reduce over M (mean pooling) into per-graph sums
    int g = blockIdx.y >> 2;   // 128 rows per block, 512 rows per graph
    #pragma unroll
    for (int j = 0; j < 4; ++j) {
        int col   = nBase + waveN * 64 + j * 16 + l16;
        float bia = bupd[col];
        float s = 0.f;
        #pragma unroll
        for (int i = 0; i < 2; ++i)
            #pragma unroll
            for (int v = 0; v < 8; ++v)
                s += fmaxf(acc[i][j][v] + bia, 0.f);
        s += __shfl_down(s, 16, 32);         // fold lane halves (same col)
        if (lane < 16)
            atomicAdd(&pooled[(size_t)g * HIDDEN + col], s);
    }
}

// ---------------------------------------------------------------- classifier head
__global__ __launch_bounds__(256) void k_cls(const float* __restrict__ pooled,
                                             const float* __restrict__ Wcls,
                                             const float* __restrict__ bcls,
                                             float* __restrict__ out) {
    int idx = blockIdx.x * 256 + threadIdx.x;   // 0..1023
    int b = idx >> 4;
    int c = idx & 15;
    float s = 0.f;
    for (int k = 0; k < HIDDEN; ++k)
        s += pooled[(size_t)b * HIDDEN + k] * Wcls[(size_t)c * HIDDEN + k];
    out[idx] = s * (1.0f / (float)NNODE) + bcls[c];
}

// ---------------------------------------------------------------- launcher
extern "C" void kernel_launch(void* const* d_in, const int* in_sizes, int n_in,
                              void* d_out, int out_size, void* d_ws, size_t ws_size,
                              hipStream_t stream) {
    const int*   tokens = (const int*)d_in[0];
    /* d_in[1] lengths: all == N, unused */
    const int*   edges  = (const int*)d_in[2];
    const float* emb    = (const float*)d_in[3];
    const float* Wmsg   = (const float*)d_in[4];
    const float* Wupd   = (const float*)d_in[5];
    const float* bupd   = (const float*)d_in[6];
    const float* Wcls   = (const float*)d_in[7];
    const float* bcls   = (const float*)d_in[8];
    float* out = (float*)d_out;

    // workspace carve-up (all offsets 256B aligned)
    char* base = (char*)d_ws;
    unsigned short* hbf   = (unsigned short*)base;                       // 32 MB
    unsigned short* wmsgb = (unsigned short*)(base + (size_t)MROWS * EMBED * 2);
    unsigned short* wupdb = (unsigned short*)((char*)wmsgb + (size_t)HIDDEN * EMBED * 2);      // +1 MB
    float*          hm    = (float*)((char*)wupdb + (size_t)HIDDEN * KTOT * 2);                // +3 MB
    unsigned short* agg   = (unsigned short*)((char*)hm + (size_t)MROWS * HIDDEN * 4);         // +128 MB
    float*          pooled= (float*)((char*)agg + (size_t)MROWS * HIDDEN * 2);                 // +64 MB

    hipMemsetAsync(agg,    0, (size_t)MROWS * HIDDEN * 2, stream);
    hipMemsetAsync(pooled, 0, (size_t)GB * HIDDEN * 4, stream);

    k_embed   <<<MROWS, 128, 0, stream>>>(tokens, emb, hbf);
    k_cvt_bf16<<<(HIDDEN * EMBED / 4) / 256, 256, 0, stream>>>(Wmsg, wmsgb, HIDDEN * EMBED / 4);
    k_cvt_bf16<<<(HIDDEN * KTOT / 4) / 256, 256, 0, stream>>>(Wupd, wupdb, HIDDEN * KTOT / 4);

    k_gemm1   <<<dim3(HIDDEN / 128, MROWS / 128), 256, 0, stream>>>(hbf, wmsgb, hm);
    k_scatter <<<GB * NEDGE, 256, 0, stream>>>(hm, edges, agg);
    k_gemm2   <<<dim3(HIDDEN / 128, MROWS / 128), 256, 0, stream>>>(hbf, agg, wupdb, bupd, pooled);
    k_cls     <<<4, 256, 0, stream>>>(pooled, Wcls, bcls, out);
}
